// IntraMetapathAggregation_2662879723953
// MI455X (gfx1250) — matrix operand order repaired and checked by reading, not compiled, not run on hardware
//
#include <hip/hip_runtime.h>
#include <hip/hip_bf16.h>

typedef __attribute__((ext_vector_type(16))) _Float16 v16h;
typedef __attribute__((ext_vector_type(8)))  _Float16 v8h;
typedef __attribute__((ext_vector_type(4)))  _Float16 v4h;
typedef __attribute__((ext_vector_type(8)))  float    v8f;
typedef __attribute__((ext_vector_type(4)))  float    v4f;

#define NNODES 100000
#define NEDGES 300000
#define NBATCH 50000
#define DEMB   128
#define DHID   64
#define NHEADS 8
#define ROWF   264      // padded LDS row stride in f16 elems (256 + 8)

// ---- monotone float<->uint key for atomicMax-based segment max ----
__device__ __forceinline__ unsigned fkey(float f) {
  unsigned u = __float_as_uint(f);
  return (u & 0x80000000u) ? ~u : (u | 0x80000000u);
}
__device__ __forceinline__ float unkey(unsigned k) {
  unsigned u = (k & 0x80000000u) ? (k & 0x7FFFFFFFu) : ~k;
  return __uint_as_float(u);
}

// ---- init: zero node_ft / den / gate, set maxkey to key(-inf) ----
__global__ void k_init(float* __restrict__ node_ft, float* __restrict__ den,
                       unsigned* __restrict__ maxkey, float* __restrict__ gate) {
  size_t i = (size_t)blockIdx.x * blockDim.x + threadIdx.x;
  if (i < (size_t)NNODES * 512) node_ft[i] = 0.f;
  if (i < (size_t)NNODES * 8)  { den[i] = 0.f; maxkey[i] = 0x007FFFFFu; /* key(-inf) */ }
  if (i < 8) gate[i] = 0.f;
}

// ---- pre-swizzle combined weight Wc[256][64] (k-major) into per-lane WMMA
//      B-fragments: addr = ((t*4+c)*32 + lane)*8 + v, value = {Wc[k0][n],Wc[k0+1][n]} f16x2
__global__ void k_wfrag(const float* __restrict__ Wi, const float* __restrict__ Wp,
                        unsigned* __restrict__ wfrag) {
  int tid = blockIdx.x * blockDim.x + threadIdx.x;
  if (tid >= 8 * 4 * 32 * 8) return;
  int v    = tid & 7;
  int lane = (tid >> 3) & 31;
  int c    = (tid >> 8) & 3;
  int t    = tid >> 10;
  int n    = c * 16 + (lane & 15);
  int kb   = t * 32 + ((lane & 16) ? 16 : 0);
  int k0   = kb + 2 * v;
  float f0 = (k0     < 128) ? Wi[n * 128 + k0]       : Wp[n * 128 + (k0 - 128)];
  float f1 = (k0 + 1 < 128) ? Wi[n * 128 + (k0 + 1)] : Wp[n * 128 + (k0 + 1 - 128)];
  union { _Float16 h[2]; unsigned u; } pk;
  pk.h[0] = (_Float16)f0;
  pk.h[1] = (_Float16)f1;
  wfrag[tid] = pk.u;
}

// ---- edge-feature GEMM via WMMA: one wave per 16-edge tile ----
__global__ __launch_bounds__(32)
void k_edata(const int* __restrict__ mpi, const float* __restrict__ tf,
             const unsigned* __restrict__ wfrag, float* __restrict__ edata) {
  __shared__ _Float16 sA[16][ROWF];
  const int lane = threadIdx.x;
  const int e0 = blockIdx.x * 16;

  // Stage A: sA[m][0..127]=f16(tf[i0]+tf[i3]); sA[m][128..255]=f16(tf[i1]+tf[i2]).
  // 32 lanes cover 128 floats as v4f chunks -> fully coalesced 512B row reads.
  for (int m = 0; m < 16; ++m) {
    int e = e0 + m;
    int i0 = mpi[e * 4 + 0], i1 = mpi[e * 4 + 1];
    int i2 = mpi[e * 4 + 2], i3 = mpi[e * 4 + 3];
    v4f a = ((const v4f*)(tf + (size_t)i0 * DEMB))[lane];
    v4f b = ((const v4f*)(tf + (size_t)i3 * DEMB))[lane];
    v4f c = ((const v4f*)(tf + (size_t)i1 * DEMB))[lane];
    v4f d = ((const v4f*)(tf + (size_t)i2 * DEMB))[lane];
    v4f si = a + b, sp = c + d;
    v4h hi_, hp_;
    #pragma unroll
    for (int j = 0; j < 4; ++j) { hi_[j] = (_Float16)si[j]; hp_[j] = (_Float16)sp[j]; }
    *(v4h*)&sA[m][lane * 4]        = hi_;
    *(v4h*)&sA[m][128 + lane * 4]  = hp_;
  }
  __syncthreads();  // single-wave WG: compiler fence + in-order LDS

  v8f acc[4] = {};
  const int m     = lane & 15;
  const int khalf = (lane < 16) ? 0 : 16;
  for (int t = 0; t < 8; ++t) {
    int kk = t * 32 + khalf;
    // A fragment in contiguous K order (matches wfrag B): VGPR0-3 = K kk..kk+7, VGPR4-7 = K kk+8..kk+15
    v8h alo = *(const v8h*)&sA[m][kk];
    v8h ahi = *(const v8h*)&sA[m][kk + 8];
    v16h afrag = __builtin_shufflevector(alo, ahi,
                   0, 1, 2, 3, 4, 5, 6, 7, 8, 9, 10, 11, 12, 13, 14, 15);
    #pragma unroll
    for (int c = 0; c < 4; ++c) {
      union { uint4 q[2]; v16h h; } bu;
      const uint4* pb = (const uint4*)(wfrag + (size_t)(((t * 4 + c) * 32 + lane) * 8));
      bu.q[0] = pb[0]; bu.q[1] = pb[1];
      acc[c] = __builtin_amdgcn_wmma_f32_16x16x32_f16(
                 false, afrag, false, bu.h, (short)0, acc[c], false, false);
    }
  }
  // C layout: VGPR r -> M=r (lanes 0-15) / M=r+8 (lanes 16-31); N = lane%16 within tile
  const int mb = (lane < 16) ? 0 : 8;
  const int nl = lane & 15;
  #pragma unroll
  for (int c = 0; c < 4; ++c)
    #pragma unroll
    for (int r = 0; r < 8; ++r)
      edata[(size_t)(e0 + mb + r) * DHID + c * 16 + nl] = acc[c][r];
}

// ---- per-edge head logits, talking-heads, leaky-relu, segment max ----
__global__ void k_logits(const float* __restrict__ edata, const int* __restrict__ dst,
                         const float* __restrict__ attn, const float* __restrict__ Wth,
                         float* __restrict__ logits, unsigned* __restrict__ maxkey) {
  int e = blockIdx.x * blockDim.x + threadIdx.x;
  if (e >= NEDGES) return;
  const float* row = edata + (size_t)e * DHID;
  float s[8] = {};
  for (int k = 0; k < DHID; ++k) {
    float x = row[k];
    #pragma unroll
    for (int h = 0; h < 8; ++h) s[h] += x * attn[h * DHID + k];  // attn uniform -> s_loads
  }
  int d = dst[e];
  #pragma unroll
  for (int h = 0; h < 8; ++h) {
    float l = 0.f;
    #pragma unroll
    for (int j = 0; j < 8; ++j) l += Wth[h * 8 + j] * s[j];
    l = (l > 0.f) ? l : 0.01f * l;
    logits[(size_t)e * 8 + h] = l;
    atomicMax(&maxkey[(size_t)d * 8 + h], fkey(l));
  }
}

// ---- exp(logit - segmax), accumulate softmax denominator ----
__global__ void k_expden(const int* __restrict__ dst, const unsigned* __restrict__ maxkey,
                         float* __restrict__ logits, float* __restrict__ den) {
  int e = blockIdx.x * blockDim.x + threadIdx.x;
  if (e >= NEDGES) return;
  int d = dst[e];
  #pragma unroll
  for (int h = 0; h < 8; ++h) {
    float mx = unkey(maxkey[(size_t)d * 8 + h]);
    float ex = __expf(logits[(size_t)e * 8 + h] - mx);
    logits[(size_t)e * 8 + h] = ex;
    atomicAdd(&den[(size_t)d * 8 + h], ex);
  }
}

// ---- message scatter: wave per edge, lane = feature index -> coalesced atomics ----
__global__ __launch_bounds__(256)
void k_scatter(const float* __restrict__ edata, const float* __restrict__ ex,
               const float* __restrict__ den, const int* __restrict__ dst,
               float* __restrict__ node_ft) {
  int gid  = blockIdx.x * 256 + threadIdx.x;
  int e    = gid >> 5;
  int lane = gid & 31;
  if (e >= NEDGES) return;
  int d = dst[e];
  float x0 = edata[(size_t)e * 64 + lane];
  float x1 = edata[(size_t)e * 64 + 32 + lane];
  float* base = node_ft + (size_t)d * 512;
  #pragma unroll
  for (int h = 0; h < 8; ++h) {
    float a = ex[(size_t)e * 8 + h] / den[(size_t)d * 8 + h];
    atomicAdd(base + h * 64 + lane,       x0 * a);
    atomicAdd(base + h * 64 + 32 + lane,  x1 * a);
  }
}

// ---- gate accumulation: sum_b dot(result[b,h,:], W_gate) ----
__global__ void k_gate(const int* __restrict__ batch, const float* __restrict__ node_ft,
                       const float* __restrict__ Wg, float* __restrict__ gate_acc) {
  int t = blockIdx.x * blockDim.x + threadIdx.x;
  if (t >= NBATCH * 8) return;
  int b = t >> 3, h = t & 7;
  const float* row = node_ft + (size_t)batch[b] * 512 + h * 64;
  float s = 0.f;
  for (int k = 0; k < 64; ++k) s += row[k] * Wg[k];
  atomicAdd(&gate_acc[h], s);
}

// ---- gather batch rows, scale by gate ----
__global__ void k_out(const int* __restrict__ batch, const float* __restrict__ node_ft,
                      const float* __restrict__ gate_acc, const float* __restrict__ bg,
                      float* __restrict__ out) {
  size_t i = (size_t)blockIdx.x * blockDim.x + threadIdx.x;
  if (i >= (size_t)NBATCH * 512) return;
  int b = (int)(i >> 9);
  int r = (int)(i & 511);
  int h = r >> 6;
  float g = gate_acc[h] * (1.0f / NBATCH) + bg[0];
  out[i] = node_ft[(size_t)batch[b] * 512 + r] * g;
}

extern "C" void kernel_launch(void* const* d_in, const int* in_sizes, int n_in,
                              void* d_out, int out_size, void* d_ws, size_t ws_size,
                              hipStream_t stream) {
  (void)in_sizes; (void)n_in; (void)out_size; (void)ws_size;
  const int*   batch = (const int*)d_in[0];
  const int*   mpi   = (const int*)d_in[1];
  const int*   edst  = (const int*)d_in[2];
  const float* tf    = (const float*)d_in[3];
  const float* Wi    = (const float*)d_in[4];
  const float* Wp    = (const float*)d_in[5];
  const float* attn  = (const float*)d_in[6];
  const float* Wth   = (const float*)d_in[7];
  const float* Wg    = (const float*)d_in[8];
  const float* bg    = (const float*)d_in[9];
  float* out = (float*)d_out;

  // workspace layout (all 16B-aligned)
  char* ws = (char*)d_ws;
  size_t off = 0;
  unsigned* wfrag  = (unsigned*)(ws + off); off += (size_t)8 * 4 * 32 * 8 * 4;     // 32 KB
  float*    edata  = (float*)(ws + off);    off += (size_t)NEDGES * 64 * 4;         // 76.8 MB
  float*    logit  = (float*)(ws + off);    off += (size_t)NEDGES * 8 * 4;          // 9.6 MB
  unsigned* maxkey = (unsigned*)(ws + off); off += (size_t)NNODES * 8 * 4;          // 3.2 MB
  float*    den    = (float*)(ws + off);    off += (size_t)NNODES * 8 * 4;          // 3.2 MB
  float*    nft    = (float*)(ws + off);    off += (size_t)NNODES * 512 * 4;        // 204.8 MB
  float*    gate   = (float*)(ws + off);    off += 64;

  k_wfrag<<<32, 256, 0, stream>>>(Wi, Wp, wfrag);
  {
    size_t total = (size_t)NNODES * 512;
    int blocks = (int)((total + 255) / 256);
    k_init<<<blocks, 256, 0, stream>>>(nft, den, maxkey, gate);
  }
  k_edata<<<NEDGES / 16, 32, 0, stream>>>(mpi, tf, wfrag, edata);        // 18750 tiles
  k_logits<<<(NEDGES + 255) / 256, 256, 0, stream>>>(edata, edst, attn, Wth, logit, maxkey);
  k_expden<<<(NEDGES + 255) / 256, 256, 0, stream>>>(edst, maxkey, logit, den);
  k_scatter<<<(NEDGES * 32) / 256, 256, 0, stream>>>(edata, logit, den, edst, nft);
  k_gate<<<(NBATCH * 8 + 255) / 256, 256, 0, stream>>>(batch, nft, Wg, gate);
  {
    size_t total = (size_t)NBATCH * 512;
    int blocks = (int)((total + 255) / 256);
    k_out<<<blocks, 256, 0, stream>>>(batch, nft, gate, bg, out);
  }
}
